// UDP_Encoder_9569187135580
// MI455X (gfx1250) — matrix operand order repaired and checked by reading, not compiled, not run on hardware
//
#include <hip/hip_runtime.h>

// ---------------- CDNA5 wave32 WMMA types ----------------
typedef __attribute__((ext_vector_type(16))) __bf16          v16bf;
typedef __attribute__((ext_vector_type(8)))  float           v8f;
typedef __attribute__((ext_vector_type(8)))  unsigned short  v8us;

#define LEAKY_SLOPE 0.01f
#define N_ENV   64
#define EMB_DIM 32
#define FEAT    128
#define HID     256
#define KPAD0   160   // IN_DIM=145 padded to multiple of 32

static __device__ __forceinline__ unsigned short f2bf(float f) {
  unsigned int u = __float_as_uint(f);
  u += 0x7FFFu + ((u >> 16) & 1u);       // round-to-nearest-even
  return (unsigned short)(u >> 16);
}
static __device__ __forceinline__ float bf2f(unsigned short h) {
  return __uint_as_float((unsigned int)h << 16);
}

struct Pair16 { v8us a, b; };
// Build a 16xbf16 WMMA fragment from two contiguous 16B segments.
static __device__ __forceinline__ v16bf load_frag(const unsigned short* p0,
                                                  const unsigned short* p1) {
  Pair16 t;
  t.a = *(const v8us*)p0;
  t.b = *(const v8us*)p1;
  return __builtin_bit_cast(v16bf, t);
}

// ---------------- build X = [obs, obs2-obs, act, rew, pad] as bf16 ----------------
__global__ void build_x(const float* __restrict__ obs, const float* __restrict__ act,
                        const float* __restrict__ obs2, const float* __restrict__ rew,
                        unsigned short* __restrict__ X, int B) {
  int t = blockIdx.x * blockDim.x + threadIdx.x;
  if (t >= B * KPAD0) return;
  int b = t / KPAD0, c = t % KPAD0;
  float v = 0.0f;
  if (c < 64)        v = obs[b * 64 + c];
  else if (c < 128)  v = obs2[b * 64 + (c - 64)] - obs[b * 64 + (c - 64)];
  else if (c < 144)  v = act[b * 16 + (c - 128)];
  else if (c == 144) v = rew[b];
  X[t] = f2bf(v);
}

// ---------------- pack a weight matrix into WMMA B-fragment order ----------------
// B-fragment (16-bit, 32x16): lane L holds column N = L&15, elements j=0..15 are
// K = 16*(L>>4) + j (this code's contiguous k order, matched by the A fragment loads).
// dst element index t = ((kt*n_tiles + nt)*32 + lane)*16 + j  -> one contiguous
// 32B chunk per lane per fragment (coalesced 1KB fragment loads in the GEMM).
__global__ void pack_b(const float* __restrict__ src, unsigned short* __restrict__ dst,
                       int Ksrc, int Ntot, int n_tiles, int total, int transposed) {
  int t = blockIdx.x * blockDim.x + threadIdx.x;
  if (t >= total) return;
  int j    = t & 15;
  int lane = (t >> 4) & 31;
  int frag = t >> 9;
  int nt   = frag % n_tiles;
  int kt   = frag / n_tiles;
  int K = kt * 32 + ((lane >> 4) * 16) + j;
  int N = nt * 16 + (lane & 15);
  float v = 0.0f;
  if (K < Ksrc)
    v = transposed ? src[(size_t)N * Ksrc + K] : src[(size_t)K * Ntot + N];
  dst[t] = f2bf(v);
}

// ---------------- generic bf16 WMMA GEMM + bias + LeakyReLU ----------------
// Block = 256 threads = 8 waves arranged 4(M) x 2(N); block tile 128x64,
// wave tile 32x32 (2x2 accumulators). A is row-major bf16 (pitch = K).
__global__ __launch_bounds__(256) void gemm_bias_leaky(
    const unsigned short* __restrict__ A, const unsigned short* __restrict__ Bp,
    const float* __restrict__ bias, unsigned short* __restrict__ C,
    int K, int N, int n_tiles_total) {
  const int lane = threadIdx.x & 31;
  const int w    = threadIdx.x >> 5;
  const int lh   = lane >> 4;       // half-wave id
  const int l15  = lane & 15;
  const int rowBase = blockIdx.x * 128 + (w >> 1) * 32;
  const int colBase = blockIdx.y * 64 + (w & 1) * 32;
  const int nt0 = colBase >> 4;

  const unsigned short* arow0 = A + (size_t)(rowBase + l15) * K;
  const unsigned short* arow1 = A + (size_t)(rowBase + 16 + l15) * K;

  v8f acc[2][2] = {};
  const int ksteps = K >> 5;
  for (int kt = 0; kt < ksteps; ++kt) {
    const int koff = kt * 32 + lh * 16;
    // A fragments (this code's contiguous k order, as in packed B: lane holds K in {koff..koff+15})
    v16bf af[2];
    af[0] = load_frag(arow0 + koff, arow0 + koff + 8);
    af[1] = load_frag(arow1 + koff, arow1 + koff + 8);
    // prefetch next k-tile of A (lowers to global_prefetch_b8)
    if (kt + 1 < ksteps) {
      __builtin_prefetch(arow0 + koff + 32, 0, 3);
      __builtin_prefetch(arow1 + koff + 32, 0, 3);
    }
    // B fragments: one contiguous 32B per lane from the packed layout
    v16bf bf[2];
    {
      const unsigned short* bp0 =
          Bp + ((size_t)(kt * n_tiles_total + nt0) * 32 + lane) * 16;
      bf[0] = load_frag(bp0, bp0 + 8);
      bf[1] = load_frag(bp0 + 512, bp0 + 520);
    }
#pragma unroll
    for (int i = 0; i < 2; ++i)
#pragma unroll
      for (int j = 0; j < 2; ++j)
        acc[i][j] = __builtin_amdgcn_wmma_f32_16x16x32_bf16(
            false, af[i], false, bf[j], (short)0, acc[i][j], false, false);
  }
  // epilogue: C tile layout -> lane L: row = r + 8*(L>>4), col = L&15
#pragma unroll
  for (int i = 0; i < 2; ++i) {
#pragma unroll
    for (int j = 0; j < 2; ++j) {
      const int col = colBase + j * 16 + l15;
      const float bv = bias[col];
#pragma unroll
      for (int r = 0; r < 8; ++r) {
        const int row = rowBase + i * 16 + r + lh * 8;
        float v = acc[i][j][r] + bv;
        v = v > 0.0f ? v : LEAKY_SLOPE * v;
        C[(size_t)row * N + col] = f2bf(v);
      }
    }
  }
}

// ---------------- fused projection GEMM + RBF distance ----------------
// One wave per (32-row tile, env): f2e tile (32x32) stays in registers
// (1:1 fragment-load : WMMA ratio, halves W_proj L2 re-reads vs 16-row tiles),
// reduce squared diff over EMB via half-wave butterfly, write distance only.
__global__ __launch_bounds__(256) void dist_kernel(
    const unsigned short* __restrict__ F, const unsigned short* __restrict__ Wpp,
    const float* __restrict__ e_emb, const float* __restrict__ sigma,
    float* __restrict__ dist) {
  const int lane = threadIdx.x & 31;
  const int gw   = blockIdx.x * 8 + (threadIdx.x >> 5);
  const int env  = gw & (N_ENV - 1);
  const int m0   = (gw >> 6) * 32;
  const int lh   = lane >> 4;
  const int l15  = lane & 15;

  const unsigned short* frow0 = F + (size_t)(m0 + l15) * FEAT;
  const unsigned short* frow1 = F + (size_t)(m0 + 16 + l15) * FEAT;

  v8f acc[2][2] = {};
#pragma unroll
  for (int kt = 0; kt < FEAT / 32; ++kt) {
    const int koff = kt * 32 + lh * 16;
    v16bf af[2];
    af[0] = load_frag(frow0 + koff, frow0 + koff + 8);
    af[1] = load_frag(frow1 + koff, frow1 + koff + 8);
    const unsigned short* bp0 =
        Wpp + ((size_t)(kt * 128 + env * 2) * 32 + lane) * 16;  // 128 n_tiles total
    v16bf bf[2];
    bf[0] = load_frag(bp0, bp0 + 8);
    bf[1] = load_frag(bp0 + 512, bp0 + 520);
#pragma unroll
    for (int i = 0; i < 2; ++i)
#pragma unroll
      for (int j = 0; j < 2; ++j)
        acc[i][j] = __builtin_amdgcn_wmma_f32_16x16x32_bf16(
            false, af[i], false, bf[j], (short)0, acc[i][j], false, false);
  }
  // squared diff vs e_emb; emb dim = l15 (tile j=0) and 16+l15 (tile j=1)
  const float em0 = e_emb[env * EMB_DIM + l15];
  const float em1 = e_emb[env * EMB_DIM + 16 + l15];
  float d2[2][8];
#pragma unroll
  for (int i = 0; i < 2; ++i)
#pragma unroll
    for (int r = 0; r < 8; ++r) {
      float x0 = acc[i][0][r] - em0, x1 = acc[i][1][r] - em1;
      d2[i][r] = x0 * x0 + x1 * x1;
    }
  // butterfly sum over the 16 lanes of each half-wave (the 16 emb columns)
#pragma unroll
  for (int m = 1; m <= 8; m <<= 1)
#pragma unroll
    for (int i = 0; i < 2; ++i)
#pragma unroll
      for (int r = 0; r < 8; ++r) d2[i][r] += __shfl_xor(d2[i][r], m, 32);

  if (l15 == 0) {
    const float s = sigma[env];
    const float inv = 1.0f / (2.0f * s * s);
#pragma unroll
    for (int i = 0; i < 2; ++i)
#pragma unroll
      for (int r = 0; r < 8; ++r) {
        const int row = m0 + i * 16 + r + lh * 8;
        dist[(size_t)row * N_ENV + env] = expf(-(d2[i][r] * (1.0f / EMB_DIM)) * inv);
      }
  }
}

// ---------------- argmin + gathers + recompute chosen embedding ----------------
// One wave per sample; first-occurrence argmin like jnp.argmin.
__global__ __launch_bounds__(256) void finalize(
    const float* __restrict__ dist, const unsigned short* __restrict__ F,
    const float* __restrict__ W_proj, const float* __restrict__ e_emb,
    float* __restrict__ ce, float* __restrict__ cd,
    float* __restrict__ idxf, float* __restrict__ cm) {
  const int lane = threadIdx.x & 31;
  const int b = blockIdx.x * 8 + (threadIdx.x >> 5);

  float v0 = dist[(size_t)b * N_ENV + lane];
  float v1 = dist[(size_t)b * N_ENV + 32 + lane];
  float bv = v0; int bi = lane;
  if (v1 < bv) { bv = v1; bi = lane + 32; }
#pragma unroll
  for (int m = 16; m >= 1; m >>= 1) {
    float ov = __shfl_xor(bv, m, 32);
    int   oi = __shfl_xor(bi, m, 32);
    if (ov < bv || (ov == bv && oi < bi)) { bv = ov; bi = oi; }
  }
  // lane == embedding dim e: recompute f2e[b, bi, e] = F[b,:] . W_proj[bi, e, :]
  const float* wrow = W_proj + ((size_t)bi * EMB_DIM + lane) * FEAT;
  const unsigned short* frow = F + (size_t)b * FEAT;
  float s = 0.0f;
  for (int f = 0; f < FEAT; ++f) s = fmaf(bf2f(frow[f]), wrow[f], s);
  ce[(size_t)b * EMB_DIM + lane] = s;
  cm[(size_t)b * EMB_DIM + lane] = e_emb[(size_t)bi * EMB_DIM + lane];
  if (lane == 0) { cd[b] = bv; idxf[b] = (float)bi; }
}

// ---------------- host side ----------------
extern "C" void kernel_launch(void* const* d_in, const int* in_sizes, int n_in,
                              void* d_out, int out_size, void* d_ws, size_t ws_size,
                              hipStream_t stream) {
  const float* obs   = (const float*)d_in[0];
  const float* act   = (const float*)d_in[1];
  const float* obs2  = (const float*)d_in[2];
  const float* rew   = (const float*)d_in[3];
  const float* W0    = (const float*)d_in[4];
  const float* b0    = (const float*)d_in[5];
  const float* W1    = (const float*)d_in[6];
  const float* b1    = (const float*)d_in[7];
  const float* W2    = (const float*)d_in[8];
  const float* b2    = (const float*)d_in[9];
  const float* Wproj = (const float*)d_in[10];
  const float* eemb  = (const float*)d_in[11];
  const float* sigma = (const float*)d_in[12];
  const int B = in_sizes[0] / 64;  // obs is (B, 64)

  // workspace carve-out (256B aligned)
  char* ws = (char*)d_ws;
  size_t off = 0;
  auto carve = [&](size_t bytes) -> char* {
    char* p = ws + off;
    off = (off + bytes + 255) & ~(size_t)255;
    return p;
  };
  unsigned short* Xbf = (unsigned short*)carve((size_t)B * KPAD0 * 2);
  unsigned short* H0  = (unsigned short*)carve((size_t)B * HID * 2);
  unsigned short* H1  = (unsigned short*)carve((size_t)B * HID * 2);
  unsigned short* Fbf = (unsigned short*)carve((size_t)B * FEAT * 2);
  unsigned short* W0p = (unsigned short*)carve((size_t)(KPAD0/32) * (HID/16) * 1024);
  unsigned short* W1p = (unsigned short*)carve((size_t)(HID/32) * (HID/16) * 1024);
  unsigned short* W2p = (unsigned short*)carve((size_t)(HID/32) * (FEAT/16) * 1024);
  unsigned short* Wpp = (unsigned short*)carve((size_t)(FEAT/32) * (N_ENV*EMB_DIM/16) * 1024);

  // output layout: ce(B*32) | cd(B) | idx(B) | cm(B*32) | dist(B*64)
  float* out  = (float*)d_out;
  float* ce   = out;
  float* cd   = out + (size_t)B * 32;
  float* idxf = out + (size_t)B * 33;
  float* cm   = out + (size_t)B * 34;
  float* dist = out + (size_t)B * 66;

  // 1) build padded bf16 input
  {
    int total = B * KPAD0;
    build_x<<<(total + 255) / 256, 256, 0, stream>>>(obs, act, obs2, rew, Xbf, B);
  }
  // 2) pack weights into WMMA B-fragment layout
  {
    int t0 = (KPAD0/32) * (HID/16) * 512;
    pack_b<<<(t0 + 255) / 256, 256, 0, stream>>>(W0, W0p, 145, HID, HID/16, t0, 0);
    int t1 = (HID/32) * (HID/16) * 512;
    pack_b<<<(t1 + 255) / 256, 256, 0, stream>>>(W1, W1p, HID, HID, HID/16, t1, 0);
    int t2 = (HID/32) * (FEAT/16) * 512;
    pack_b<<<(t2 + 255) / 256, 256, 0, stream>>>(W2, W2p, HID, FEAT, FEAT/16, t2, 0);
    // W_proj[n_env][emb][feat] -> B[k=feat][n=env*32+emb] (transposed source)
    int t3 = (FEAT/32) * (N_ENV*EMB_DIM/16) * 512;
    pack_b<<<(t3 + 255) / 256, 256, 0, stream>>>(Wproj, Wpp, FEAT, N_ENV*EMB_DIM,
                                                 N_ENV*EMB_DIM/16, t3, 1);
  }
  // 3) three MLP GEMMs (bias + leaky fused)
  gemm_bias_leaky<<<dim3(B / 128, HID / 64), 256, 0, stream>>>(
      Xbf, W0p, b0, H0, KPAD0, HID, HID / 16);
  gemm_bias_leaky<<<dim3(B / 128, HID / 64), 256, 0, stream>>>(
      H0, W1p, b1, H1, HID, HID, HID / 16);
  gemm_bias_leaky<<<dim3(B / 128, FEAT / 64), 256, 0, stream>>>(
      H1, W2p, b2, Fbf, HID, FEAT, FEAT / 16);
  // 4) fused projection + RBF distance (no f2e materialization: saves ~1GB HBM traffic)
  dist_kernel<<<(B / 32) * N_ENV / 8, 256, 0, stream>>>(Fbf, Wpp, eemb, sigma, dist);
  // 5) argmin + gathers + recompute chosen embedding
  finalize<<<B / 8, 256, 0, stream>>>(dist, Fbf, Wproj, eemb, ce, cd, idxf, cm);
}
